// M2Model_69449621176665
// MI455X (gfx1250) — matrix operand order repaired and checked by reading, not compiled, not run on hardware
//
#include <hip/hip_runtime.h>
#include <hip/hip_bf16.h>

// ---------------------------------------------------------------------------
// Bi-directional Mamba2 block for MI455X (gfx1250, wave32, WMMA).
// All five GEMMs run on v_wmma_f32_16x16x32_f16 (f16 operands, f32 accum),
// register-blocked 1(A-frag) x 4(B-frag) per wave so the inner loop is
// 4 WMMA per 5 fragment loads with zero exec-mask churn.
// Sequential SSM scan / conv / norm stay in f32 VALU (tiny FLOP count,
// precision-critical, latency-bound).
// ---------------------------------------------------------------------------

typedef __attribute__((ext_vector_type(16))) _Float16 v16h;
typedef __attribute__((ext_vector_type(8)))  _Float16 v8h;
typedef __attribute__((ext_vector_type(8)))  float    v8f;

#define M_ROWS    8192      // 32 * 256 flattened (bm, L)
#define D_MODEL   512
#define D_INNER   1024
#define D_STATE   32
#define NHEADS    16
#define HEADDIM   64
#define CONV_DIM  1088      // D_INNER + 2*D_STATE
#define D_IN_PROJ 2128      // 2*D_INNER + 2*D_STATE + NHEADS
#define SEQ_L     256
#define N_BM      32
#define VOCAB_N   28

// ---------------------------------------------------------------------------
// WMMA GEMM:  C[M,N] = A[M,K] * W[N,K]^T (+ bias[N])
//
// Fragment layouts (CDNA5 16x16x32 f16, wave32):
//  A (16x32 MxK): lane&15 = row; lane>>4 picks K {0..15} vs
//                 {16..31} -> two contiguous 16B loads.
//  B (32x16 KxN) fed as W[N,K] row-major: lane&15 = col; lane>>4 picks
//                 K {0..15} vs {16..31} -> one contiguous 32B load.
//  C/D: VGPR v -> row v + 8*(lane>>4), col = lane&15.
//
// Wave tile: 16(M) x 64(N) = 4 WMMA tiles sharing one A fragment.
// Block: 256 threads = 8 waves as 4(M) x 2(N) -> 64(M) x 128(N) per block.
// Out-of-range B rows are CLAMPED (loads always legal, no exec divergence);
// their columns are masked only at the final store.  M, K multiples of 32.
// ---------------------------------------------------------------------------
__global__ __launch_bounds__(256) void wmma_gemm_kernel(
    const _Float16* __restrict__ A, const _Float16* __restrict__ W,
    float* __restrict__ C, const float* __restrict__ bias,
    int M, int N, int K) {
  const int lane = threadIdx.x & 31;
  const int wave = threadIdx.x >> 5;
  const int hs   = lane >> 4;     // K-half select
  const int rr   = lane & 15;     // row (A) / col (B) within tile
  const int m0 = blockIdx.y * 64 + (wave >> 1) * 16;
  const int n0 = blockIdx.x * 128 + (wave & 1) * 64;

  // A fragment base: row m0+rr, element offset hs*16 (second 16B chunk at +8)
  const _Float16* __restrict__ Ab = A + (size_t)(m0 + rr) * K + hs * 16;

  // B fragment bases, one per 16-col tile; clamp rows so loads stay in-bounds
  const _Float16* __restrict__ Wb[4];
#pragma unroll
  for (int j = 0; j < 4; ++j) {
    int br = n0 + j * 16 + rr;
    if (br >= N) br = N - 1;              // clamped columns never stored
    Wb[j] = W + (size_t)br * K + hs * 16;
  }

  v8f acc[4] = {};
  for (int k0 = 0; k0 < K; k0 += 32) {
    v8h lo = *(const v8h*)(Ab + k0);
    v8h hi = *(const v8h*)(Ab + k0 + 8);
    v16h a;
#pragma unroll
    for (int i = 0; i < 8; ++i) { a[i] = lo[i]; a[8 + i] = hi[i]; }
#pragma unroll
    for (int j = 0; j < 4; ++j) {
      v16h b = *(const v16h*)(Wb[j] + k0);
      acc[j] = __builtin_amdgcn_wmma_f32_16x16x32_f16(
          /*neg_a=*/false, a, /*neg_b=*/false, b,
          /*c_mod=*/(short)0, acc[j], /*reuse_a=*/false, /*reuse_b=*/false);
    }
  }

#pragma unroll
  for (int j = 0; j < 4; ++j) {
    const int col = n0 + j * 16 + rr;
    if (col < N) {
      float bv = bias ? bias[col] : 0.0f;
#pragma unroll
      for (int v = 0; v < 8; ++v) {
        const int row = m0 + v + 8 * hs;
        C[(size_t)row * N + col] = acc[j][v] + bv;
      }
    }
  }
}

// ---------------------------------------------------------------------------
// Embedding gather: X rows are r = bm*256 + l ; token = inputs[l, bm] with
// inputs shaped (BATCH=256, STEPS=32).  Emits f32 (residual), f16 (GEMM A),
// and L-flipped f16 (backward direction A).
// ---------------------------------------------------------------------------
__global__ void embed_kernel(const int* __restrict__ inputs,
                             const float* __restrict__ emb,
                             float* __restrict__ Xf32,
                             _Float16* __restrict__ Xf16,
                             _Float16* __restrict__ Xrf16) {
  const int r  = blockIdx.x;
  const int bm = r >> 8, l = r & 255;
  const int tok = inputs[l * 32 + bm];
  const int rf  = bm * 256 + (255 - l);
  for (int d = threadIdx.x; d < D_MODEL; d += blockDim.x) {
    float v = emb[(size_t)tok * D_MODEL + d];
    Xf32[(size_t)r * D_MODEL + d] = v;
    _Float16 h = (_Float16)v;
    Xf16 [(size_t)r  * D_MODEL + d] = h;
    Xrf16[(size_t)rf * D_MODEL + d] = h;
  }
}

__global__ void f32_to_f16_kernel(const float* __restrict__ src,
                                  _Float16* __restrict__ dst, int n) {
  int i = blockIdx.x * blockDim.x + threadIdx.x;
  if (i < n) dst[i] = (_Float16)src[i];
}

// ---------------------------------------------------------------------------
// dt = softplus(zxbcdt[:, 2112+h] + dt_bias[h]);  dA = exp(dt * (-exp(A_log)))
// ---------------------------------------------------------------------------
__global__ void dtda_kernel(const float* __restrict__ zxb,
                            const float* __restrict__ dt_bias,
                            const float* __restrict__ A_log,
                            float* __restrict__ dt, float* __restrict__ dA) {
  int i = blockIdx.x * blockDim.x + threadIdx.x;
  if (i >= M_ROWS * NHEADS) return;
  int h = i & 15, r = i >> 4;
  float raw = zxb[(size_t)r * D_IN_PROJ + (D_IN_PROJ - NHEADS) + h] + dt_bias[h];
  float d = (raw > 20.0f) ? raw : log1pf(expf(raw));
  float a = -expf(A_log[h]);
  dt[i] = d;
  dA[i] = expf(d * a);
}

// ---------------------------------------------------------------------------
// Depthwise causal conv (k=4) + SiLU along L, rolling window per channel.
// Reads the xBC slice (cols 1024..2111) of zxbcdt, writes (32,256,1088).
// Coalesced across channels at each l.
// ---------------------------------------------------------------------------
__global__ void conv_kernel(const float* __restrict__ zxb,
                            const float* __restrict__ conv_w,
                            const float* __restrict__ conv_b,
                            float* __restrict__ xBCact) {
  const int c  = blockIdx.y * 256 + threadIdx.x;
  const int bm = blockIdx.x;
  if (c >= CONV_DIM) return;
  const float w0 = conv_w[c * 4 + 0], w1 = conv_w[c * 4 + 1];
  const float w2 = conv_w[c * 4 + 2], w3 = conv_w[c * 4 + 3];
  const float b  = conv_b[c];
  const float* src = zxb + (size_t)bm * SEQ_L * D_IN_PROJ + D_INNER + c;
  float* dst = xBCact + (size_t)bm * SEQ_L * CONV_DIM + c;
  float x0 = 0.f, x1 = 0.f, x2 = 0.f;
  for (int l = 0; l < SEQ_L; ++l) {
    float x3 = src[(size_t)l * D_IN_PROJ];
    float s  = b + w0 * x0 + w1 * x1 + w2 * x2 + w3 * x3;
    dst[(size_t)l * CONV_DIM] = s / (1.f + expf(-s));   // SiLU
    x0 = x1; x1 = x2; x2 = x3;
  }
}

// ---------------------------------------------------------------------------
// Sequential SSM scan. One 64-thread block per (bm, head); thread p owns
// state row h[p, 0..31] in 32 VGPRs. B_t / C_t staged through LDS per step.
// Fuses the "+ x * D" skip term into y.
// ---------------------------------------------------------------------------
__global__ __launch_bounds__(64) void scan_kernel(
    const float* __restrict__ xBCact, const float* __restrict__ dt,
    const float* __restrict__ dA, const float* __restrict__ Dp,
    float* __restrict__ yscan) {
  const int bm = blockIdx.x >> 4;
  const int h  = blockIdx.x & 15;
  const int p  = threadIdx.x;           // 0..63
  __shared__ float Bs[D_STATE], Cs[D_STATE];

  float s[D_STATE];
#pragma unroll
  for (int n = 0; n < D_STATE; ++n) s[n] = 0.f;
  const float Dh = Dp[h];
  const float* base = xBCact + (size_t)bm * SEQ_L * CONV_DIM;

  for (int l = 0; l < SEQ_L; ++l) {
    const float* rowp = base + (size_t)l * CONV_DIM;
    if (p < D_STATE) Bs[p]            = rowp[D_INNER + p];
    else             Cs[p - D_STATE]  = rowp[D_INNER + D_STATE + (p - D_STATE)];
    __syncthreads();

    const int ri   = (bm * SEQ_L + l) * NHEADS + h;
    const float dAv  = dA[ri];
    const float xv   = rowp[h * HEADDIM + p];
    const float coef = dt[ri] * xv;
    float y = xv * Dh;
#pragma unroll
    for (int n = 0; n < D_STATE; ++n) {
      s[n] = s[n] * dAv + coef * Bs[n];
      y   += s[n] * Cs[n];
    }
    yscan[(size_t)(bm * SEQ_L + l) * D_INNER + h * HEADDIM + p] = y;
    __syncthreads();
  }
}

// ---------------------------------------------------------------------------
// y *= silu(z); RMS-norm over 1024; * norm_w; emit f16 for out_proj GEMM.
// ---------------------------------------------------------------------------
__global__ __launch_bounds__(256) void gatenorm_kernel(
    const float* __restrict__ zxb, const float* __restrict__ yscan,
    const float* __restrict__ norm_w, _Float16* __restrict__ yn16) {
  const int r = blockIdx.x;
  __shared__ float red[256];
  float y[4];
  float acc = 0.f;
#pragma unroll
  for (int j = 0; j < 4; ++j) {
    int c = threadIdx.x + j * 256;
    float z = zxb[(size_t)r * D_IN_PROJ + c];
    float g = z / (1.f + expf(-z));
    float v = yscan[(size_t)r * D_INNER + c] * g;
    y[j] = v;
    acc += v * v;
  }
  red[threadIdx.x] = acc;
  __syncthreads();
  for (int off = 128; off > 0; off >>= 1) {
    if (threadIdx.x < off) red[threadIdx.x] += red[threadIdx.x + off];
    __syncthreads();
  }
  const float scale = rsqrtf(red[0] * (1.f / D_INNER) + 1e-5f);
#pragma unroll
  for (int j = 0; j < 4; ++j) {
    int c = threadIdx.x + j * 256;
    yn16[(size_t)r * D_INNER + c] = (_Float16)(y[j] * scale * norm_w[c]);
  }
}

// ---------------------------------------------------------------------------
// Y = concat(yf + X, flip_L(yb) + X) -> f16 for the head GEMM.
// ---------------------------------------------------------------------------
__global__ void combine_kernel(const float* __restrict__ yf,
                               const float* __restrict__ yb,
                               const float* __restrict__ Xf32,
                               _Float16* __restrict__ Y16) {
  const int r  = blockIdx.x;
  const int bm = r >> 8, l = r & 255;
  const int rb = bm * 256 + (255 - l);
  for (int c = threadIdx.x; c < D_MODEL; c += blockDim.x) {
    float xv = Xf32[(size_t)r * D_MODEL + c];
    Y16[(size_t)r * D_INNER + c] =
        (_Float16)(yf[(size_t)r * D_MODEL + c] + xv);
    Y16[(size_t)r * D_INNER + D_MODEL + c] =
        (_Float16)(yb[(size_t)rb * D_MODEL + c] + xv);
  }
}

// ---------------------------------------------------------------------------
// Host launch sequence
// ---------------------------------------------------------------------------
extern "C" void kernel_launch(void* const* d_in, const int* in_sizes, int n_in,
                              void* d_out, int out_size, void* d_ws, size_t ws_size,
                              hipStream_t stream) {
  (void)in_sizes; (void)n_in; (void)out_size; (void)ws_size;

  const int*   inputs = (const int*)d_in[0];
  const float* emb    = (const float*)d_in[2];
  const float* in_proj_w[2]  = {(const float*)d_in[3],  (const float*)d_in[11]};
  const float* conv_w[2]     = {(const float*)d_in[4],  (const float*)d_in[12]};
  const float* conv_b[2]     = {(const float*)d_in[5],  (const float*)d_in[13]};
  const float* dt_bias[2]    = {(const float*)d_in[6],  (const float*)d_in[14]};
  const float* A_log[2]      = {(const float*)d_in[7],  (const float*)d_in[15]};
  const float* Dp[2]         = {(const float*)d_in[8],  (const float*)d_in[16]};
  const float* norm_w[2]     = {(const float*)d_in[9],  (const float*)d_in[17]};
  const float* out_proj_w[2] = {(const float*)d_in[10], (const float*)d_in[18]};
  const float* lin_w = (const float*)d_in[19];
  const float* lin_b = (const float*)d_in[20];

  // ---- workspace carve (all 256B aligned; ~243 MB total) ----
  char*  ws  = (char*)d_ws;
  size_t off = 0;
  auto carve = [&](size_t bytes) -> void* {
    void* p = ws + off;
    off = (off + bytes + 255) & ~(size_t)255;
    return p;
  };
  float*    Xf32   = (float*)   carve((size_t)M_ROWS * D_MODEL * 4);
  _Float16* Xf16   = (_Float16*)carve((size_t)M_ROWS * D_MODEL * 2);
  _Float16* Xrf16  = (_Float16*)carve((size_t)M_ROWS * D_MODEL * 2);
  _Float16* wbuf16 = (_Float16*)carve((size_t)D_IN_PROJ * D_MODEL * 2); // largest weight
  _Float16* linw16 = (_Float16*)carve((size_t)VOCAB_N * D_INNER * 2);
  float*    zxb    = (float*)   carve((size_t)M_ROWS * D_IN_PROJ * 4);
  float*    dtb    = (float*)   carve((size_t)M_ROWS * NHEADS * 4);
  float*    dAb    = (float*)   carve((size_t)M_ROWS * NHEADS * 4);
  float*    xBCact = (float*)   carve((size_t)M_ROWS * CONV_DIM * 4);
  float*    yscan  = (float*)   carve((size_t)M_ROWS * D_INNER * 4);
  _Float16* yn16   = (_Float16*)carve((size_t)M_ROWS * D_INNER * 2);
  float*    dirout[2];
  dirout[0]        = (float*)   carve((size_t)M_ROWS * D_MODEL * 4);
  dirout[1]        = (float*)   carve((size_t)M_ROWS * D_MODEL * 4);
  _Float16* Y16    = (_Float16*)carve((size_t)M_ROWS * D_INNER * 2);

  // 1) embedding gather (+f16 and flipped-f16 copies)
  embed_kernel<<<M_ROWS, 256, 0, stream>>>(inputs, emb, Xf32, Xf16, Xrf16);

  // 2) per direction
  for (int d = 0; d < 2; ++d) {
    int n = D_IN_PROJ * D_MODEL;
    f32_to_f16_kernel<<<(n + 255) / 256, 256, 0, stream>>>(in_proj_w[d], wbuf16, n);

    // in_proj GEMM: (8192x512) x (2128x512)^T  -> zxb
    {
      dim3 grid((D_IN_PROJ + 127) / 128, M_ROWS / 64);
      wmma_gemm_kernel<<<grid, 256, 0, stream>>>(
          (d == 0) ? Xf16 : Xrf16, wbuf16, zxb, nullptr,
          M_ROWS, D_IN_PROJ, D_MODEL);
    }

    dtda_kernel<<<(M_ROWS * NHEADS + 255) / 256, 256, 0, stream>>>(
        zxb, dt_bias[d], A_log[d], dtb, dAb);

    conv_kernel<<<dim3(N_BM, (CONV_DIM + 255) / 256), 256, 0, stream>>>(
        zxb, conv_w[d], conv_b[d], xBCact);

    scan_kernel<<<N_BM * NHEADS, 64, 0, stream>>>(xBCact, dtb, dAb, Dp[d], yscan);

    gatenorm_kernel<<<M_ROWS, 256, 0, stream>>>(zxb, yscan, norm_w[d], yn16);

    n = D_MODEL * D_INNER;
    f32_to_f16_kernel<<<(n + 255) / 256, 256, 0, stream>>>(out_proj_w[d], wbuf16, n);

    // out_proj GEMM: (8192x1024) x (512x1024)^T -> dirout[d]
    {
      dim3 grid((D_MODEL + 127) / 128, M_ROWS / 64);
      wmma_gemm_kernel<<<grid, 256, 0, stream>>>(
          yn16, wbuf16, dirout[d], nullptr, M_ROWS, D_MODEL, D_INNER);
    }
  }

  // 3) residual + concat (un-flips the backward direction)
  combine_kernel<<<M_ROWS, 256, 0, stream>>>(dirout[0], dirout[1], Xf32, Y16);

  // 4) head GEMM with fused bias: (8192x1024) x (28x1024)^T + lin_b -> d_out
  {
    int n = VOCAB_N * D_INNER;
    f32_to_f16_kernel<<<(n + 255) / 256, 256, 0, stream>>>(lin_w, linw16, n);
    dim3 grid((VOCAB_N + 127) / 128, M_ROWS / 64);
    wmma_gemm_kernel<<<grid, 256, 0, stream>>>(
        Y16, linw16, (float*)d_out, lin_b, M_ROWS, VOCAB_N, D_INNER);
  }
}
